// SpatialSelfAttention_59811714564809
// MI455X (gfx1250) — matrix operand order repaired and checked by reading, not compiled, not run on hardware
//
#include <hip/hip_runtime.h>

// ---------------------------------------------------------------------------
// SpatialSelfAttention for MI455X (gfx1250, wave32, WMMA bf16 16x16x32)
// ---------------------------------------------------------------------------
static constexpr int BB = 16;    // batch
static constexpr int TT = 12;    // time
static constexpr int NN = 358;   // nodes
static constexpr int DD = 128;   // model dim
static constexpr int KH = 8;     // heads
static constexpr int HS = 16;    // head size
static constexpr int NPAD = 384; // 12 * 32 (paired m-tiles)
static constexpr int QT = 23;    // ceil(358/16) q-row tiles
static constexpr int MP = 12;    // m-tile pairs (24 tiles, last fully masked)

typedef __bf16 v16bf __attribute__((ext_vector_type(16)));
typedef float  v8f   __attribute__((ext_vector_type(8)));

// A-matrix fragment element -> K index (contiguous, matches B fragments):
//  lanes 0-15: VGPR j holds K=2j,2j+1
//  lanes 16-31: +16.
__device__ __forceinline__ int a_k_of(int j, int hi) {
    return hi * 16 + 2 * j;
}

// Load A fragment from a row-major matrix row.  IT = float or __bf16.
template <typename IT>
__device__ __forceinline__ v16bf load_a_row(const IT* rowptr, int kbase, int hi) {
    v16bf af;
#pragma unroll
    for (int j = 0; j < 8; ++j) {
        int k0 = kbase + a_k_of(j, hi);
        if constexpr (sizeof(IT) == 2) {
            union { unsigned u; __bf16 h[2]; } cv;
            cv.u = *(const unsigned*)(rowptr + k0);
            af[2 * j] = cv.h[0]; af[2 * j + 1] = cv.h[1];
        } else {
            float2 f = *(const float2*)(rowptr + k0);
            af[2 * j] = (__bf16)f.x; af[2 * j + 1] = (__bf16)f.y;
        }
    }
    return af;
}

// B fragment, K-contiguous source row: element e -> k = kbase + hi*16 + e.
// p points at the column's K-vector (stride 1 in k).
template <typename IT>
__device__ __forceinline__ v16bf load_b_row(const IT* p, int kbase, int hi) {
    v16bf bf_;
    const IT* q = p + kbase + hi * 16;
#pragma unroll
    for (int j = 0; j < 8; ++j) {
        if constexpr (sizeof(IT) == 2) {
            union { unsigned u; __bf16 h[2]; } cv;
            cv.u = *(const unsigned*)(q + 2 * j);
            bf_[2 * j] = cv.h[0]; bf_[2 * j + 1] = cv.h[1];
        } else {
            float2 f = *(const float2*)(q + 2 * j);
            bf_[2 * j] = (__bf16)f.x; bf_[2 * j + 1] = (__bf16)f.y;
        }
    }
    return bf_;
}

// K^T B-fragment for S = Q*K^T (K-dim = h = 16, upper half zero).
__device__ __forceinline__ v16bf load_kT_frag(const __bf16* Ksh_, int mt, int n0, int hi) {
    v16bf kf;
    if (hi == 0) {
        const unsigned* kp = (const unsigned*)(Ksh_ + (mt * 16 + n0) * HS);
#pragma unroll
        for (int j = 0; j < 8; ++j) {
            union { unsigned u; __bf16 h[2]; } cv; cv.u = kp[j];
            kf[2 * j] = cv.h[0]; kf[2 * j + 1] = cv.h[1];
        }
    } else {
#pragma unroll
        for (int e = 0; e < 16; ++e) kf[e] = (__bf16)0.f;
    }
    return kf;
}

__device__ __forceinline__ void wave_lds_sync() {
    __builtin_amdgcn_wave_barrier();
    asm volatile("s_wait_dscnt 0x0" ::: "memory");
    __builtin_amdgcn_wave_barrier();
}

// ---------------------------------------------------------------------------
// Kernel A: fused per-(b,t,head) KV projection + flash attention.
// grid = B*T*KH blocks, 256 threads (8 waves).
// ---------------------------------------------------------------------------
template <typename OT>
__global__ __launch_bounds__(256) void attn_kernel(
    const float* __restrict__ x,  const float* __restrict__ wk,
    const float* __restrict__ bk, const float* __restrict__ wv,
    const float* __restrict__ bv, OT* __restrict__ obuf) {
    __shared__ __bf16 Qsh[NPAD * HS];
    __shared__ __bf16 Ksh[NPAD * HS];
    __shared__ __bf16 Vsh[NPAD * HS];
    __shared__ __bf16 WkT[HS * DD];       // wk head slice, [col][k]
    __shared__ __bf16 WvT[HS * DD];       // wv head slice, [col][k]
    __shared__ float  Psc[8 * 16 * 32];   // per-wave P-transpose scratch

    const int bid = blockIdx.x;
    const int kh  = bid % KH;
    const int t   = (bid / KH) % TT;
    const int b   = bid / (KH * TT);
    const int tid = threadIdx.x, lane = tid & 31, wid = tid >> 5;
    const int hi = lane >> 4, n0 = lane & 15;

    const float* xbt = x + (size_t)(b * TT + t) * NN * DD;  // [N][D]
    const float* wkb = wk + (size_t)b * DD * DD;
    const float* wvb = wv + (size_t)b * DD * DD;

    // zero only the node-pad region [NN, NPAD) of Q/K/V
    for (int i = tid + NN * HS; i < NPAD * HS; i += 256) {
        Qsh[i] = (__bf16)0.f; Ksh[i] = (__bf16)0.f; Vsh[i] = (__bf16)0.f;
    }
    // Q head slice straight from x
    for (int i = tid; i < NN * HS; i += 256) {
        int node = i >> 4, hh = i & 15;
        Qsh[i] = (__bf16)xbt[node * DD + kh * HS + hh];
    }
    // weight head slices, transposed to [col][k] (K-contiguous for B frags)
    for (int i = tid; i < HS * DD; i += 256) {
        int col = i & 15, k = i >> 4;
        WkT[col * DD + k] = (__bf16)wkb[(size_t)k * DD + kh * HS + col];
        WvT[col * DD + k] = (__bf16)wvb[(size_t)k * DD + kh * HS + col];
    }
    __syncthreads();

    // ---- stage 1: K/V head slices, K_blk = x @ wk[:, kh*16:+16] + bk ----
    for (int rt = wid; rt < QT; rt += 8) {
        v8f ck = {}, cvv = {};
        int arow = rt * 16 + n0;                // A lane row (node)
        if (arow >= NN) arow = NN - 1;          // clamp loads; stores guarded
        const float* ap = xbt + (size_t)arow * DD;
#pragma unroll
        for (int kc = 0; kc < 4; ++kc) {
            v16bf af  = load_a_row<float>(ap, kc * 32, hi);
            v16bf bkf = load_b_row<__bf16>(WkT + n0 * DD, kc * 32, hi);
            v16bf bvf = load_b_row<__bf16>(WvT + n0 * DD, kc * 32, hi);
            ck  = __builtin_amdgcn_wmma_f32_16x16x32_bf16(false, af, false, bkf, (short)0, ck,  false, false);
            cvv = __builtin_amdgcn_wmma_f32_16x16x32_bf16(false, af, false, bvf, (short)0, cvv, false, false);
        }
        float bkv = bk[b * DD + kh * HS + n0];
        float bvv = bv[b * DD + kh * HS + n0];
#pragma unroll
        for (int r = 0; r < 8; ++r) {
            int node = rt * 16 + r + 8 * hi;
            if (node < NN) {
                Ksh[node * HS + n0] = (__bf16)(ck[r]  + bkv);
                Vsh[node * HS + n0] = (__bf16)(cvv[r] + bvv);
            }
        }
    }
    __syncthreads();

    // ---- stage 2: flash attention, one wave per 16-row Q tile ----
    const float scale = 0.25f;  // 1/sqrt(h=16)
    float* psc = Psc + wid * (16 * 32);
    for (int qt = wid; qt < QT; qt += 8) {
        // Q A-fragment: k-dim = h = 16, zero-pad K=16..31
        v16bf qf;
#pragma unroll
        for (int j = 0; j < 8; ++j) {
            if (hi == 0) {
                int k0 = 2 * j;
                union { unsigned u; __bf16 h[2]; } cv;
                cv.u = *(const unsigned*)(&Qsh[(qt * 16 + n0) * HS + k0]);
                qf[2 * j] = cv.h[0]; qf[2 * j + 1] = cv.h[1];
            } else {
                qf[2 * j] = (__bf16)0.f; qf[2 * j + 1] = (__bf16)0.f;
            }
        }

        v8f acc = {};
        float mrun[8], lrun[8];
#pragma unroll
        for (int r = 0; r < 8; ++r) { mrun[r] = -1e30f; lrun[r] = 0.f; }

        for (int mp = 0; mp < MP; ++mp) {       // 32 nodes per iteration
            const int mt0 = 2 * mp, mt1 = 2 * mp + 1;
            v16bf kf0 = load_kT_frag(Ksh, mt0, n0, hi);
            v16bf kf1 = load_kT_frag(Ksh, mt1, n0, hi);
            v8f zc = {};
            v8f c0 = __builtin_amdgcn_wmma_f32_16x16x32_bf16(false, qf, false, kf0, (short)0, zc, false, false);
            v8f c1 = __builtin_amdgcn_wmma_f32_16x16x32_bf16(false, qf, false, kf1, (short)0, zc, false, false);

            // scale + column mask
            const bool v0 = (mt0 * 16 + n0) < NN;
            const bool v1 = (mt1 * 16 + n0) < NN;
#pragma unroll
            for (int r = 0; r < 8; ++r) {
                c0[r] = v0 ? c0[r] * scale : -1e30f;
                c1[r] = v1 ? c1[r] * scale : -1e30f;
            }

            // row-max across the 16 lanes of each half-wave, over both tiles
            float bm[8];
#pragma unroll
            for (int r = 0; r < 8; ++r) bm[r] = fmaxf(c0[r], c1[r]);
            for (int off = 1; off < 16; off <<= 1)
#pragma unroll
                for (int r = 0; r < 8; ++r) bm[r] = fmaxf(bm[r], __shfl_xor(bm[r], off, 32));

            float p0[8], p1[8], rs[8];
#pragma unroll
            for (int r = 0; r < 8; ++r) {
                float mnew = fmaxf(mrun[r], bm[r]);
                float sc = __expf(mrun[r] - mnew);
                p0[r] = __expf(c0[r] - mnew);
                p1[r] = __expf(c1[r] - mnew);
                rs[r] = p0[r] + p1[r];
                acc[r] *= sc;
                lrun[r] *= sc;
                mrun[r] = mnew;
            }
            for (int off = 1; off < 16; off <<= 1)
#pragma unroll
                for (int r = 0; r < 8; ++r) rs[r] += __shfl_xor(rs[r], off, 32);
#pragma unroll
            for (int r = 0; r < 8; ++r) lrun[r] += rs[r];

            // transpose P (C layout -> A layout) through per-wave LDS scratch
#pragma unroll
            for (int r = 0; r < 8; ++r) {
                psc[(r + 8 * hi) * 32 + n0]      = p0[r];
                psc[(r + 8 * hi) * 32 + 16 + n0] = p1[r];
            }
            wave_lds_sync();
            v16bf pf;                 // fully dense A fragment, K = 32 nodes
#pragma unroll
            for (int j = 0; j < 8; ++j) {
                int k0 = a_k_of(j, hi);
                pf[2 * j]     = (__bf16)psc[n0 * 32 + k0];
                pf[2 * j + 1] = (__bf16)psc[n0 * 32 + k0 + 1];
            }
            v16bf vf;                 // fully dense B fragment: V[mp*32+k][n]
#pragma unroll
            for (int e = 0; e < 16; ++e)
                vf[e] = Vsh[(mp * 32 + hi * 16 + e) * HS + n0];
            acc = __builtin_amdgcn_wmma_f32_16x16x32_bf16(false, pf, false, vf, (short)0, acc, false, false);
            __builtin_amdgcn_wave_barrier();   // keep next-iter psc stores below the reads
        }

        // epilogue: normalize + store head slice of o
#pragma unroll
        for (int r = 0; r < 8; ++r) {
            float l = lrun[r];
            float o = acc[r] / (l > 0.f ? l : 1.f);
            int node = qt * 16 + r + 8 * hi;
            if (node < NN)
                obuf[((size_t)(b * TT + t) * NN + node) * DD + kh * HS + n0] = (OT)o;
        }
    }
}

// ---------------------------------------------------------------------------
// Kernel B: out = relu(o @ W1^T + b1) @ W2^T + b2.
// 128 threads = 4 waves, each wave one 16-row tile (rows divide evenly).
// o tile is DMA'd to LDS with async global->LDS (ASYNCcnt path).
// ---------------------------------------------------------------------------
template <typename IT>
__global__ __launch_bounds__(128) void proj_kernel(
    const IT* __restrict__ obuf, const float* __restrict__ W1,
    const float* __restrict__ b1, const float* __restrict__ W2,
    const float* __restrict__ b2, float* __restrict__ out) {
    constexpr int TILEB = 16 * DD * (int)sizeof(IT);        // bytes per wave tile
    __shared__ __bf16 t1sh[4 * 16 * DD];
    __shared__ alignas(16) unsigned char otile[4 * TILEB];

    const int tid = threadIdx.x, lane = tid & 31, wid = tid >> 5;
    const int hi = lane >> 4, n0 = lane & 15;
    const size_t row0 = ((size_t)blockIdx.x * 4 + wid) * 16;
    __bf16* t1 = t1sh + wid * 16 * DD;
    unsigned char* mytile = otile + wid * TILEB;

    // async DMA: 16x128 o tile -> LDS (b128 per lane, ASYNCcnt tracked)
    {
        const unsigned char* gsrc = (const unsigned char*)(obuf + row0 * DD);
#pragma unroll
        for (int it = 0; it < TILEB / 512; ++it) {
            int li = it * 32 + lane;
            unsigned lo = (unsigned)(uintptr_t)(mytile + li * 16);
            unsigned long long ga = (unsigned long long)(uintptr_t)(gsrc + li * 16);
            asm volatile("global_load_async_to_lds_b128 %0, %1, off"
                         :: "v"(lo), "v"(ga) : "memory");
        }
    }

    // prefetch the stage-2 weights while the DMA flies
    __builtin_prefetch((const void*)(W2 + (size_t)(lane * 4) * DD), 0, 3);

    asm volatile("s_wait_asynccnt 0x0" ::: "memory");
    __builtin_amdgcn_wave_barrier();
    const IT* orows = (const IT*)mytile;                    // [16][128]

    // ---- stage 1: t1 = relu(o @ W1^T + b1) ----
    v8f acc[8];
#pragma unroll
    for (int nc = 0; nc < 8; ++nc) acc[nc] = (v8f){};
#pragma unroll
    for (int kc = 0; kc < 4; ++kc) {
        v16bf af = load_a_row<IT>(orows + n0 * DD, kc * 32, hi);
#pragma unroll
        for (int nc = 0; nc < 8; ++nc) {
            v16bf bf_ = load_b_row<float>(W1 + (size_t)(nc * 16 + n0) * DD, kc * 32, hi);
            acc[nc] = __builtin_amdgcn_wmma_f32_16x16x32_bf16(false, af, false, bf_, (short)0, acc[nc], false, false);
        }
    }
#pragma unroll
    for (int nc = 0; nc < 8; ++nc) {
        float bias = b1[nc * 16 + n0];
#pragma unroll
        for (int r = 0; r < 8; ++r) {
            float v = acc[nc][r] + bias;
            t1[(r + 8 * hi) * DD + nc * 16 + n0] = (__bf16)(v > 0.f ? v : 0.f);
        }
    }
    wave_lds_sync();

    // ---- stage 2: out = t1 @ W2^T + b2 ----
    v8f acc2[8];
#pragma unroll
    for (int nc = 0; nc < 8; ++nc) acc2[nc] = (v8f){};
#pragma unroll
    for (int kc = 0; kc < 4; ++kc) {
        v16bf af = load_a_row<__bf16>(t1 + n0 * DD, kc * 32, hi);
#pragma unroll
        for (int nc = 0; nc < 8; ++nc) {
            v16bf bf_ = load_b_row<float>(W2 + (size_t)(nc * 16 + n0) * DD, kc * 32, hi);
            acc2[nc] = __builtin_amdgcn_wmma_f32_16x16x32_bf16(false, af, false, bf_, (short)0, acc2[nc], false, false);
        }
    }
#pragma unroll
    for (int nc = 0; nc < 8; ++nc) {
        float bias = b2[nc * 16 + n0];
#pragma unroll
        for (int r = 0; r < 8; ++r)
            out[(row0 + r + 8 * hi) * DD + nc * 16 + n0] = acc2[nc][r] + bias;
    }
}

// ---------------------------------------------------------------------------
extern "C" void kernel_launch(void* const* d_in, const int* in_sizes, int n_in,
                              void* d_out, int out_size, void* d_ws, size_t ws_size,
                              hipStream_t stream) {
    const float* x  = (const float*)d_in[0];
    const float* wk = (const float*)d_in[1];
    const float* bk = (const float*)d_in[2];
    const float* wv = (const float*)d_in[3];
    const float* bv = (const float*)d_in[4];
    const float* W1 = (const float*)d_in[5];
    const float* b1 = (const float*)d_in[6];
    const float* W2 = (const float*)d_in[7];
    const float* b2 = (const float*)d_in[8];
    float* out = (float*)d_out;

    const int attn_blocks = BB * TT * KH;                  // 1536
    const int proj_blocks = (BB * TT * NN) / (4 * 16);     // 68736/64 = 1074
    const size_t obf_bytes = (size_t)BB * TT * NN * DD * 2;

    if (ws_size >= obf_bytes) {
        __bf16* obuf = (__bf16*)d_ws;
        attn_kernel<__bf16><<<attn_blocks, 256, 0, stream>>>(x, wk, bk, wv, bv, obuf);
        proj_kernel<__bf16><<<proj_blocks, 128, 0, stream>>>(obuf, W1, b1, W2, b2, out);
    } else {
        // fallback: f32 o intermediate lives in d_out; MLP is tile-local in-place
        attn_kernel<float><<<attn_blocks, 256, 0, stream>>>(x, wk, bk, wv, bv, out);
        proj_kernel<float><<<proj_blocks, 128, 0, stream>>>(out, W1, b1, W2, b2, out);
    }
}
